// Quanv_43791486550369
// MI455X (gfx1250) — matrix operand order repaired and checked by reading, not compiled, not run on hardware
//
#include <hip/hip_runtime.h>
#include <hip/hip_bf16.h>
#include <vector>
#include <cstdint>

typedef _Float16 v8h  __attribute__((ext_vector_type(8)));
typedef _Float16 v16h __attribute__((ext_vector_type(16)));
typedef float    v8f  __attribute__((ext_vector_type(8)));

// ---------------------------------------------------------------------------
// Host-side replication of numpy legacy RandomState (MT19937) so we can
// reproduce PennyLane RandomLayers' static circuit structure (seed 42).
// ---------------------------------------------------------------------------
namespace {

struct NpMT19937 {
    uint32_t mt[624];
    int mti;
    explicit NpMT19937(uint32_t s) {
        mt[0] = s;
        for (int i = 1; i < 624; i++)
            mt[i] = 1812433253u * (mt[i - 1] ^ (mt[i - 1] >> 30)) + (uint32_t)i;
        mti = 624;
    }
    uint32_t next_u32() {
        if (mti >= 624) {
            for (int i = 0; i < 624; i++) {
                uint32_t y = (mt[i] & 0x80000000u) | (mt[(i + 1) % 624] & 0x7fffffffu);
                mt[i] = mt[(i + 397) % 624] ^ (y >> 1) ^ ((y & 1u) ? 2567483615u : 0u);
            }
            mti = 0;
        }
        uint32_t y = mt[mti++];
        y ^= y >> 11;
        y ^= (y << 7) & 2636928640u;
        y ^= (y << 15) & 4022730752u;
        y ^= y >> 18;
        return y;
    }
    // numpy random_sample(): 53-bit double from two 32-bit draws
    double random_sample() {
        uint32_t a = next_u32() >> 5, b = next_u32() >> 6;
        return (a * 67108864.0 + b) / 9007199254740992.0;
    }
    // legacy rk_interval: masked rejection, 32-bit draws, result in [0, maxv]
    uint32_t bounded(uint32_t maxv) {
        if (maxv == 0) return 0;
        uint32_t mask = maxv;
        mask |= mask >> 1; mask |= mask >> 2; mask |= mask >> 4;
        mask |= mask >> 8; mask |= mask >> 16;
        uint32_t v;
        do { v = next_u32() & mask; } while (v > maxv);
        return v;
    }
};

struct GateOp {
    int type;   // 0 = rotation, 1 = cnot
    int a;      // rot: gate kind (0 RX,1 RY,2 RZ); cnot: control wire
    int b;      // rot: wire; cnot: target wire
    int widx;   // rot: flat weight index l*9+i
};

static void build_ops(std::vector<GateOp>& ops) {
    const int n_layers = 2, n_rot = 9, n_qubits = 9;
    NpMT19937 rng(42);
    for (int l = 0; l < n_layers; l++) {
        int i = 0;
        int guard = 0;
        while (i < n_rot && guard++ < 4096) {
            if (rng.random_sample() > 0.3) {              // place a rotation
                int g = (int)rng.bounded(2);              // randint(3)
                int w = (int)rng.bounded(n_qubits - 1);   // randint(9)
                ops.push_back({0, g, w, l * 9 + i});
                i++;
            } else {                                      // CNOT via permutation(9)[:2]
                int arr[9];
                for (int k = 0; k < 9; k++) arr[k] = k;
                for (int k = 8; k > 0; k--) {             // legacy Fisher-Yates
                    int j = (int)rng.bounded((uint32_t)k);
                    int t = arr[k]; arr[k] = arr[j]; arr[j] = t;
                }
                ops.push_back({1, arr[0], arr[1], 0});
            }
        }
    }
}

} // namespace

// ---------------------------------------------------------------------------
// Step 1: U = I  (512x512 complex, split Ur/Ui, row-major, f32)
// ---------------------------------------------------------------------------
__global__ void k_init_u(float* __restrict__ Ur, float* __restrict__ Ui) {
    int t = blockIdx.x * 256 + threadIdx.x;     // 262144 threads
    int r = t >> 9, c = t & 511;
    Ur[t] = (r == c) ? 1.0f : 0.0f;
    Ui[t] = 0.0f;
}

// ---------------------------------------------------------------------------
// Step 2a: apply 1-qubit rotation gate in-place:  M <- (G (x) I) M
// bitpos = 8 - wire (wire 0 is the MSB of the 9-bit state index).
// ---------------------------------------------------------------------------
__global__ void k_rot(float* __restrict__ Ur, float* __restrict__ Ui,
                      const float* __restrict__ wts, int g, int bitpos, int widx) {
    int t = blockIdx.x * 256 + threadIdx.x;     // 256 pairs * 512 cols = 131072
    int col = t & 511, pr = t >> 9;
    unsigned mask = (1u << bitpos) - 1u;
    int i0 = (int)(((unsigned)(pr & ~mask) << 1) | ((unsigned)pr & mask));
    int i1 = i0 | (1 << bitpos);

    float th = 0.5f * wts[widx];
    float sh, chf;
    __sincosf(th, &sh, &chf);

    float u00r, u00i, u01r, u01i, u10r, u10i, u11r, u11i;
    if (g == 0) {        // RX
        u00r = chf; u00i = 0.f; u01r = 0.f; u01i = -sh;
        u10r = 0.f; u10i = -sh; u11r = chf; u11i = 0.f;
    } else if (g == 1) { // RY
        u00r = chf; u00i = 0.f; u01r = -sh; u01i = 0.f;
        u10r = sh;  u10i = 0.f; u11r = chf; u11i = 0.f;
    } else {             // RZ
        u00r = chf; u00i = -sh; u01r = 0.f; u01i = 0.f;
        u10r = 0.f; u10i = 0.f; u11r = chf; u11i = sh;
    }

    int ia = i0 * 512 + col, ib = i1 * 512 + col;
    float ar = Ur[ia], ai = Ui[ia], br = Ur[ib], bi = Ui[ib];
    float nar = u00r * ar - u00i * ai + u01r * br - u01i * bi;
    float nai = u00r * ai + u00i * ar + u01r * bi + u01i * br;
    float nbr = u10r * ar - u10i * ai + u11r * br - u11i * bi;
    float nbi = u10r * ai + u10i * ar + u11r * bi + u11i * br;
    Ur[ia] = nar; Ui[ia] = nai;
    Ur[ib] = nbr; Ui[ib] = nbi;
}

// ---------------------------------------------------------------------------
// Step 2b: CNOT in-place: swap rows (c=1,t=0) <-> (c=1,t=1).
// bc/bt are bit positions (8 - wire).
// ---------------------------------------------------------------------------
__global__ void k_cnot(float* __restrict__ Ur, float* __restrict__ Ui, int bc, int bt) {
    int t = blockIdx.x * 256 + threadIdx.x;     // 128 pairs * 512 cols = 65536
    int col = t & 511, pr = t >> 9;             // 7 free bits
    int p1 = bc < bt ? bc : bt;
    int p2 = bc < bt ? bt : bc;
    int low = pr & ((1 << p1) - 1);
    int mid = (pr >> p1) & ((1 << (p2 - p1 - 1)) - 1);
    int hi  = pr >> (p2 - 1);
    int idx = (hi << (p2 + 1)) | (mid << (p1 + 1)) | low;   // zeros at bc, bt
    int r0 = idx | (1 << bc);
    int r1 = r0 | (1 << bt);
    int ia = r0 * 512 + col, ib = r1 * 512 + col;
    float tr = Ur[ia]; Ur[ia] = Ur[ib]; Ur[ib] = tr;
    float ti = Ui[ia]; Ui[ia] = Ui[ib]; Ui[ib] = ti;
}

// ---------------------------------------------------------------------------
// Step 3: pack W2 (1024x512: row 2i = Ur_i, row 2i+1 = Ui_i) into f16 in the
// exact v_wmma_f32_16x16x32_f16 A-fragment order:
//   half index o = ((kt*64 + mtile)*32 + lane)*16 + h
//   lane<16: M = mtile*16+lane,    h0..7 -> K+0..7,  h8..15 -> K+16..23
//   lane>=16: M = mtile*16+lane-16, h0..7 -> K+8..15, h8..15 -> K+24..31
// ---------------------------------------------------------------------------
__global__ void k_repack(const float* __restrict__ Ur, const float* __restrict__ Ui,
                         _Float16* __restrict__ Apack) {
    int o = blockIdx.x * 256 + threadIdx.x;     // 524288
    int h = o & 15;
    int lane = (o >> 4) & 31;
    int mtile = (o >> 9) & 63;
    int kt = o >> 15;
    int M = mtile * 16 + (lane & 15);
    int koff = (h & 7) + ((h & 8) ? 16 : 0) + ((lane & 16) ? 8 : 0);
    int K = kt * 32 + koff;
    int i = M >> 1;
    float v = (M & 1) ? Ui[i * 512 + K] : Ur[i * 512 + K];
    Apack[o] = (_Float16)v;
}

// ---------------------------------------------------------------------------
// Step 4: main fused kernel. One workgroup = 4 output pixels x 8 channels
// = 32 N-columns. GEMM [1024 x 512] x [512 x 32] with f16 WMMA, f32 acc;
// then |.|^2, Pauli-Z sign reduction, channel sum, deterministic reduce.
// ---------------------------------------------------------------------------
__global__ __launch_bounds__(256)
void quanv_main(const _Float16* __restrict__ Apack, const float* __restrict__ x,
                float* __restrict__ out) {
    __shared__ _Float16 sbuf[32 * 512];   // 32 KB: s vectors, [col][K]
    __shared__ float epart[16 * 36];      // per (wave,nt) partial e sums

    const int tid = threadIdx.x;
    const int hw0 = blockIdx.x * 4;       // 15876 / 4 = 3969 blocks, no tail

    // ---- build the 32 encoded statevectors in LDS (f16) -------------------
    {
        int slot = tid >> 3;              // 0..31 : column (c = slot>>2, dhw = slot&3)
        int part = tid & 7;               // top-3 bits of the 9-bit state index
        int c = slot >> 2, dhw = slot & 3;
        int hw = hw0 + dhw;
        int hh = hw / 126, ww = hw - hh * 126;
        const float* px = x + c * 16384 + hh * 128 + ww;
        float cs[9], sn[9];
#pragma unroll
        for (int j = 0; j < 9; j++) {
            float v = px[(j / 3) * 128 + (j % 3)];
            __sincosf(1.57079632679489662f * v, &sn[j], &cs[j]); // angle = pi*x/2
        }
        float pre = (part & 4 ? sn[0] : cs[0]) *
                    (part & 2 ? sn[1] : cs[1]) *
                    (part & 1 ? sn[2] : cs[2]);
        _Float16* dst = &sbuf[slot * 512 + part * 64];
        for (int low = 0; low < 64; low++) {
            float p = pre;
            p *= (low & 32) ? sn[3] : cs[3];
            p *= (low & 16) ? sn[4] : cs[4];
            p *= (low & 8)  ? sn[5] : cs[5];
            p *= (low & 4)  ? sn[6] : cs[6];
            p *= (low & 2)  ? sn[7] : cs[7];
            p *= (low & 1)  ? sn[8] : cs[8];
            dst[low] = (_Float16)p;
        }
    }
    __syncthreads();

    // ---- GEMM: wave w handles M rows [w*128, w*128+128), 2 N-tiles --------
    const int lane = tid & 31;
    const int wv = tid >> 5;
    const int lhi = lane >> 4;
    const int llo = lane & 15;

    v8f acc[8][2];
#pragma unroll
    for (int mt = 0; mt < 8; mt++)
#pragma unroll
        for (int nt = 0; nt < 2; nt++)
            acc[mt][nt] = (v8f){0.f, 0.f, 0.f, 0.f, 0.f, 0.f, 0.f, 0.f};

    for (int kt = 0; kt < 16; kt++) {
        v16h bfrag[2];
#pragma unroll
        for (int nt = 0; nt < 2; nt++) {
            const v8h* bp = (const v8h*)&sbuf[(nt * 16 + llo) * 512 + kt * 32 + lhi * 8];
            bfrag[nt] = __builtin_shufflevector(bp[0], bp[2],
                          0, 1, 2, 3, 4, 5, 6, 7, 8, 9, 10, 11, 12, 13, 14, 15);
        }
#pragma unroll
        for (int mt = 0; mt < 8; mt++) {
            const v8h* ap = (const v8h*)(Apack +
                              ((size_t)(((kt * 64) + (wv * 8 + mt)) * 32 + lane) << 4));
            v16h afrag = __builtin_shufflevector(ap[0], ap[1],
                          0, 1, 2, 3, 4, 5, 6, 7, 8, 9, 10, 11, 12, 13, 14, 15);
            acc[mt][0] = __builtin_amdgcn_wmma_f32_16x16x32_f16(
                             false, afrag, false, bfrag[0], (short)0, acc[mt][0], false, false);
            acc[mt][1] = __builtin_amdgcn_wmma_f32_16x16x32_f16(
                             false, afrag, false, bfrag[1], (short)0, acc[mt][1], false, false);
        }
    }

    // ---- |y|^2, Pauli-Z signed sums, deterministic reductions -------------
#pragma unroll
    for (int nt = 0; nt < 2; nt++) {
        float e[9] = {0.f, 0.f, 0.f, 0.f, 0.f, 0.f, 0.f, 0.f, 0.f};
#pragma unroll
        for (int mt = 0; mt < 8; mt++) {
#pragma unroll
            for (int j = 0; j < 4; j++) {
                float yr = acc[mt][nt][2 * j];
                float yi = acc[mt][nt][2 * j + 1];
                float p = yr * yr + yi * yi;
                int i = wv * 64 + mt * 8 + lhi * 4 + j;     // state index 0..511
#pragma unroll
                for (int ch = 0; ch < 9; ch++)
                    e[ch] += ((i >> (8 - ch)) & 1) ? -p : p;
            }
        }
        // reduce over lane bits {2,3,4}: lanes sharing (dhw, ch) within the wave
#pragma unroll
        for (int ch = 0; ch < 9; ch++) {
            float v = e[ch];
            v += __shfl_xor(v, 4, 32);
            v += __shfl_xor(v, 8, 32);
            v += __shfl_xor(v, 16, 32);
            if ((lane & 28) == 0)   // lanes 0..3 hold the wave-partial for dhw=lane
                epart[(wv * 2 + nt) * 36 + lane * 9 + ch] = v;
        }
    }
    __syncthreads();

    if (tid < 36) {     // fixed-order final sum over 16 (wave,nt) partials
        float s = 0.f;
#pragma unroll
        for (int q = 0; q < 16; q++) s += epart[q * 36 + tid];
        int dhw = tid / 9, ch = tid - dhw * 9;
        out[(size_t)(hw0 + dhw) * 9 + ch] = s;   // raw (Ho,Wo,9) reshape order
    }
}

// ---------------------------------------------------------------------------
// Launch
// ---------------------------------------------------------------------------
extern "C" void kernel_launch(void* const* d_in, const int* in_sizes, int n_in,
                              void* d_out, int out_size, void* d_ws, size_t ws_size,
                              hipStream_t stream) {
    const float* x   = (const float*)d_in[0];   // (1,8,128,128) f32
    const float* wts = (const float*)d_in[1];   // (2,9) f32
    float* out = (float*)d_out;                 // 142884 f32

    float* Ur = (float*)d_ws;                   // 512*512 f32  (1 MB)
    float* Ui = Ur + 512 * 512;                 // 512*512 f32  (1 MB)
    _Float16* Apack = (_Float16*)(Ui + 512 * 512); // 1024*512 f16 (1 MB)

    k_init_u<<<1024, 256, 0, stream>>>(Ur, Ui);

    std::vector<GateOp> ops;
    build_ops(ops);
    for (const GateOp& op : ops) {
        if (op.type == 0)
            k_rot<<<512, 256, 0, stream>>>(Ur, Ui, wts, op.a, 8 - op.b, op.widx);
        else
            k_cnot<<<256, 256, 0, stream>>>(Ur, Ui, 8 - op.a, 8 - op.b);
    }

    k_repack<<<2048, 256, 0, stream>>>(Ur, Ui, Apack);
    quanv_main<<<3969, 256, 0, stream>>>(Apack, x, out);
}
